// OnlineVideoTransformer_64570538328542
// MI455X (gfx1250) — matrix operand order repaired and checked by reading, not compiled, not run on hardware
//
#include <hip/hip_runtime.h>
#include <hip/hip_bf16.h>
#include <math.h>

// ---------------------------------------------------------------------------
// Types for CDNA5 WMMA
// ---------------------------------------------------------------------------
typedef __attribute__((ext_vector_type(16))) _Float16 v16h;
typedef __attribute__((ext_vector_type(8)))  float    v8f;

#define NB   64      // batch (BS*NW)
#define CCH  512     // channels
#define TT   128     // sequence length
#define WW   16      // window
#define NHD  8       // heads
#define HD   64      // head dim
#define MEMC 112     // history capacity (columns)
#define HEXP 2048    // MLP hidden

// ---------------------------------------------------------------------------
// f32 -> f16 conversion (weights)
// ---------------------------------------------------------------------------
__global__ __launch_bounds__(256) void cvt_f16_kernel(const float* __restrict__ src,
                                                      _Float16* __restrict__ dst, int n) {
    int i = blockIdx.x * 256 + threadIdx.x;
    if (i < n) dst[i] = (_Float16)src[i];
}

// ---------------------------------------------------------------------------
// im2col pack for k=3 pad=1 conv: x (64,512,128) f32 -> col (64,1536,128) f16
// row index r = c*3 + kk  (matches w[o][c][kk] row-major layout)
// ---------------------------------------------------------------------------
__global__ __launch_bounds__(256) void pack_conv_kernel(const float* __restrict__ x,
                                                        _Float16* __restrict__ col) {
    int idx = blockIdx.x * 256 + threadIdx.x;
    const int total = NB * 1536 * TT;
    if (idx >= total) return;
    int t = idx & (TT - 1);
    int r = (idx >> 7) % 1536;
    int b = idx / (1536 * TT);
    int c  = r / 3;
    int kk = r % 3;
    int ts = t + kk - 1;
    float v = (ts >= 0 && ts < TT) ? x[(long)b * (CCH * TT) + (long)c * TT + ts] : 0.0f;
    col[idx] = (_Float16)v;
}

// ---------------------------------------------------------------------------
// Batched WMMA GEMM:  D[b] = A (MxK, f16, row-major, shared over batch) * B[b] (KxN)
// B staged through LDS with on-the-fly f16 convert. Epilogue: 0=none 1=bias 2=bias+gelu
// Grid: (M/64, N/16, batch); block = 128 (4 waves, one 16x16 D tile each).
// B stage: thread owns column tid&15, rows (tid>>4)+{0,8,16,24} -> 4 clause-able
// global loads per K-step, pointers strength-reduced outside the loop.
// ---------------------------------------------------------------------------
#define BPITCH 48   // LDS row pitch in halves (96B -> 16B aligned rows)

template <typename BT, int EPI>
__global__ __launch_bounds__(128) void gemm_wmma_kernel(
    const _Float16* __restrict__ A, const BT* __restrict__ B,
    const float* __restrict__ bias, float* __restrict__ D,
    int M, int N, int K, int ldb, long bsb, int ldd, long bsd)
{
    __shared__ _Float16 Bs[16 * BPITCH];

    const int tid  = threadIdx.x;
    const int b    = blockIdx.z;
    const int wave = tid >> 5;
    const int lane = tid & 31;
    const int hi   = lane >> 4;
    const int nl   = lane & 15;
    const int m0   = blockIdx.x * 64 + wave * 16;
    const int n0   = blockIdx.y * 16;

    // B staging assignment (per thread): column sn, base row sk
    const int sn = tid & 15;
    const int sk = tid >> 4;                     // 0..7
    const BT* bptr = B + (long)b * bsb + n0 + (long)sk * ldb + sn;
    _Float16* bsout = &Bs[sn * BPITCH + sk];
    const long ldb8 = (long)ldb * 8;

    // A row pointer (per lane), advanced by 32 halves per K-step
    const _Float16* arow = A + (long)(m0 + nl) * K;

    v8f acc = {};

    for (int k0 = 0; k0 < K; k0 += 32) {
        // stage B panel (32 K x 16 N) -> LDS f16, layout [n][k]
        float f0 = (float)bptr[0];
        float f1 = (float)bptr[ldb8];
        float f2 = (float)bptr[ldb8 * 2];
        float f3 = (float)bptr[ldb8 * 3];
        bptr += ldb8 * 4;
        bsout[0]  = (_Float16)f0;
        bsout[8]  = (_Float16)f1;
        bsout[16] = (_Float16)f2;
        bsout[24] = (_Float16)f3;
        __syncthreads();

        union { v16h v; uint4 u[2]; } a, bb;
        // A tile: lane-lo chunks K {0-7,16-23}, lane-hi {8-15,24-31}  (ISA 16-bit A layout)
        a.u[0] = *(const uint4*)(arow + hi * 8);
        a.u[1] = *(const uint4*)(arow + 16 + hi * 8);
        arow += 32;
        // B tile: lane-lo K {0-7,16-23}, lane-hi {8-15,24-31}, N = lane&15
        const _Float16* brow = &Bs[nl * BPITCH + hi * 8];
        bb.u[0] = *(const uint4*)(brow);
        bb.u[1] = *(const uint4*)(brow + 16);

        acc = __builtin_amdgcn_wmma_f32_16x16x32_f16(
            /*neg_a=*/false, a.v, /*neg_b=*/false, bb.v,
            /*c_mod=*/(short)0, acc, /*reuse_a=*/false, /*reuse_b=*/false);
        __syncthreads();
    }

    float* Dp = D + (long)b * bsd;
#pragma unroll
    for (int i = 0; i < 8; i++) {
        int m = m0 + i + 8 * hi;      // D layout: VGPR i -> M = i (+8 for hi half)
        int n = n0 + nl;
        float v = acc[i];
        if (EPI >= 1) v += bias[m];
        if (EPI == 2) {
            float x3 = v * v * v;
            v = 0.5f * v * (1.0f + tanhf(0.7978845608028654f * (v + 0.044715f * x3)));
        }
        Dp[(long)m * ldd + n] = v;
    }
}

// ---------------------------------------------------------------------------
// Channel LayerNorm per token (mean/var over 512 channels), optional ReLU.
// Grid: (ncols, batch); block = 128, 4 channels/thread.
// ---------------------------------------------------------------------------
__global__ __launch_bounds__(128) void ln_kernel(
    const float* __restrict__ x, float* __restrict__ y,
    const float* __restrict__ g, const float* __restrict__ bt,
    int ldx, long bsx, int ldy, long bsy, int relu)
{
    const int col = blockIdx.x;
    const int b   = blockIdx.y;
    const int tid = threadIdx.x;
    const float* xp = x + (long)b * bsx + col;
    float*       yp = y + (long)b * bsy + col;

    float v[4], s = 0.f, s2 = 0.f;
#pragma unroll
    for (int i = 0; i < 4; i++) {
        int c = tid + i * 128;
        float t = xp[(long)c * ldx];
        v[i] = t; s += t; s2 += t * t;
    }
    __shared__ float r1[128], r2[128];
    r1[tid] = s; r2[tid] = s2;
    __syncthreads();
    for (int off = 64; off > 0; off >>= 1) {
        if (tid < off) { r1[tid] += r1[tid + off]; r2[tid] += r2[tid + off]; }
        __syncthreads();
    }
    float mu   = r1[0] * (1.0f / CCH);
    float var  = r2[0] * (1.0f / CCH) - mu * mu;
    float rinv = rsqrtf(var + 1e-5f);
#pragma unroll
    for (int i = 0; i < 4; i++) {
        int c = tid + i * 128;
        float o = (v[i] - mu) * rinv * g[c] + bt[c];
        if (relu) o = fmaxf(o, 0.0f);
        yp[(long)c * ldy] = o;
    }
}

// ---------------------------------------------------------------------------
// Attention core per (head, batch): scores (16 x TK), softmax, P*V.
// q/kc/vc: (64,512,16); kh/vh: (64,512,112) with Lh valid cols. o: (64,512,16)
// ---------------------------------------------------------------------------
__global__ __launch_bounds__(256) void attn_kernel(
    const float* __restrict__ q,  const float* __restrict__ kc,
    const float* __restrict__ vc, const float* __restrict__ kh,
    const float* __restrict__ vh, float* __restrict__ o, int Lh)
{
    const int h = blockIdx.x, b = blockIdx.y, tid = threadIdx.x;
    const int TK = WW + Lh;
    const int dbase = h * HD;
    const long bs16 = (long)CCH * WW;    // 8192
    const long bsh  = (long)CCH * MEMC;  // 57344

    __shared__ float qs[HD][WW + 1];
    __shared__ float S[WW][TT + 1];

    for (int i = tid; i < HD * WW; i += 256) {
        int d = i >> 4, qi = i & 15;
        qs[d][qi] = q[(long)b * bs16 + (long)(dbase + d) * WW + qi];
    }
    __syncthreads();

    for (int i = tid; i < WW * TK; i += 256) {
        int ki = i / WW, qi = i % WW;
        float acc = 0.f;
        if (ki < WW) {
            const float* kp = kc + (long)b * bs16 + (long)dbase * WW + ki;
            for (int d = 0; d < HD; d++) acc += qs[d][qi] * kp[(long)d * WW];
        } else {
            const float* kp = kh + (long)b * bsh + (long)dbase * MEMC + (ki - WW);
            for (int d = 0; d < HD; d++) acc += qs[d][qi] * kp[(long)d * MEMC];
        }
        S[qi][ki] = acc * 0.125f;   // (q*scale)·(k*scale), scale^2 = 1/sqrt(64)
    }
    __syncthreads();

    if (tid < WW) {
        float mx = -1e30f;
        for (int ki = 0; ki < TK; ki++) mx = fmaxf(mx, S[tid][ki]);
        float sum = 0.f;
        for (int ki = 0; ki < TK; ki++) { float e = expf(S[tid][ki] - mx); S[tid][ki] = e; sum += e; }
        float inv = 1.0f / sum;
        for (int ki = 0; ki < TK; ki++) S[tid][ki] *= inv;
    }
    __syncthreads();

    for (int i = tid; i < HD * WW; i += 256) {
        int d = i >> 4, qi = i & 15;
        float acc = 0.f;
        const float* vp = vc + (long)b * bs16 + (long)(dbase + d) * WW;
        for (int ki = 0; ki < WW; ki++) acc += S[qi][ki] * vp[ki];
        if (Lh > 0) {
            const float* vph = vh + (long)b * bsh + (long)(dbase + d) * MEMC;
            for (int ki = 0; ki < Lh; ki++) acc += S[qi][WW + ki] * vph[ki];
        }
        o[(long)b * bs16 + (long)(dbase + d) * WW + qi] = acc;
    }
}

// ---------------------------------------------------------------------------
// Layer-scale residual:  x[b,c,col] += ls[c] * h[b,c,col]   (16 cols)
// ---------------------------------------------------------------------------
__global__ __launch_bounds__(256) void resid_kernel(
    float* __restrict__ x, const float* __restrict__ h, const float* __restrict__ ls)
{
    int idx = blockIdx.x * 256 + threadIdx.x;
    const int total = NB * CCH * WW;
    if (idx >= total) return;
    int col = idx & 15;
    int c   = (idx >> 4) & (CCH - 1);
    int b   = idx >> 13;
    x[(long)b * (CCH * TT) + (long)c * TT + col] +=
        ls[c] * h[(long)b * (CCH * WW) + (long)c * WW + col];
}

// ---------------------------------------------------------------------------
// Append current window (slice of main buffer) into per-layer history
// ---------------------------------------------------------------------------
__global__ __launch_bounds__(256) void append_hist_kernel(
    const float* __restrict__ cur, float* __restrict__ hist, int coloff)
{
    int idx = blockIdx.x * 256 + threadIdx.x;
    const int total = NB * CCH * WW;
    if (idx >= total) return;
    int col = idx & 15;
    int c   = (idx >> 4) & (CCH - 1);
    int b   = idx >> 13;
    hist[(long)b * (CCH * MEMC) + (long)c * MEMC + coloff + col] =
        cur[(long)b * (CCH * TT) + (long)c * TT + col];
}

// ---------------------------------------------------------------------------
// Host orchestration
// ---------------------------------------------------------------------------
// params flatten order (JAX sorted-key pytree): d_in[0]=x, d_in[1]=mask,
// 2=conv0_w 3=conv1_w 4=fc_b 5=fc_w, 6..149 = 8 layers x 18 leaves, 150..153 = norm0/1.
enum { L_B1 = 0, L_B2, L_BK, L_BO, L_BQ, L_BV, L_LN1B, L_LN1G, L_LN2B, L_LN2G,
       L_LS1, L_LS2, L_W1, L_W2, L_WK, L_WO, L_WQ, L_WV };

static inline void cvt(const float* s, _Float16* d, int n, hipStream_t st) {
    cvt_f16_kernel<<<(n + 255) / 256, 256, 0, st>>>(s, d, n);
}

extern "C" void kernel_launch(void* const* d_in, const int* in_sizes, int n_in,
                              void* d_out, int out_size, void* d_ws, size_t ws_size,
                              hipStream_t stream) {
    (void)in_sizes; (void)n_in; (void)out_size; (void)ws_size;

    const float* x_in = (const float*)d_in[0];
    auto P = [&](int i) { return (const float*)d_in[i]; };
    auto L = [&](int li, int j) { return (const float*)d_in[6 + li * 18 + j]; };

    // ---- workspace carve (256B aligned) ----
    char* wsp = (char*)d_ws;
    auto alloc = [&](size_t bytes) { void* p = (void*)wsp; wsp += (bytes + 255) & ~(size_t)255; return p; };

    _Float16* fcw_h   = (_Float16*)alloc((size_t)CCH * 2048 * 2);
    _Float16* conv_h0 = (_Float16*)alloc((size_t)CCH * 1536 * 2);
    _Float16* conv_h1 = (_Float16*)alloc((size_t)CCH * 1536 * 2);
    _Float16 *wq_h[8], *wk_h[8], *wv_h[8], *wo_h[8], *w1_h[8], *w2_h[8];
    for (int li = 0; li < 8; li++) {
        wq_h[li] = (_Float16*)alloc((size_t)CCH * CCH * 2);
        wk_h[li] = (_Float16*)alloc((size_t)CCH * CCH * 2);
        wv_h[li] = (_Float16*)alloc((size_t)CCH * CCH * 2);
        wo_h[li] = (_Float16*)alloc((size_t)CCH * CCH * 2);
        w1_h[li] = (_Float16*)alloc((size_t)HEXP * CCH * 2);
        w2_h[li] = (_Float16*)alloc((size_t)CCH * HEXP * 2);
    }
    float*    XA     = (float*)alloc((size_t)NB * CCH * TT * 4);
    float*    XB     = (float*)alloc((size_t)NB * CCH * TT * 4);
    _Float16* colbuf = (_Float16*)alloc((size_t)NB * 1536 * TT * 2);
    float* hist[8];
    for (int li = 0; li < 8; li++) hist[li] = (float*)alloc((size_t)NB * CCH * MEMC * 4);
    float* hln = (float*)alloc((size_t)NB * CCH * MEMC * 4);
    float* khb = (float*)alloc((size_t)NB * CCH * MEMC * 4);
    float* vhb = (float*)alloc((size_t)NB * CCH * MEMC * 4);
    float* bl  = (float*)alloc((size_t)NB * CCH * WW * 4);
    float* bq_ = (float*)alloc((size_t)NB * CCH * WW * 4);
    float* bk_ = (float*)alloc((size_t)NB * CCH * WW * 4);
    float* bv_ = (float*)alloc((size_t)NB * CCH * WW * 4);
    float* bo_ = (float*)alloc((size_t)NB * CCH * WW * 4);
    float* bh_ = (float*)alloc((size_t)NB * CCH * WW * 4);
    float* bg  = (float*)alloc((size_t)NB * HEXP * WW * 4);

    // ---- weight conversion to f16 ----
    cvt(P(5), fcw_h,   CCH * 2048, stream);
    cvt(P(2), conv_h0, CCH * 1536, stream);
    cvt(P(3), conv_h1, CCH * 1536, stream);
    for (int li = 0; li < 8; li++) {
        cvt(L(li, L_WQ), wq_h[li], CCH * CCH,  stream);
        cvt(L(li, L_WK), wk_h[li], CCH * CCH,  stream);
        cvt(L(li, L_WV), wv_h[li], CCH * CCH,  stream);
        cvt(L(li, L_WO), wo_h[li], CCH * CCH,  stream);
        cvt(L(li, L_W1), w1_h[li], HEXP * CCH, stream);
        cvt(L(li, L_W2), w2_h[li], CCH * HEXP, stream);
    }

    const long bsX = (long)CCH * TT;     // 65536
    const long bsW = (long)CCH * WW;     // 8192
    const long bsH = (long)CCH * MEMC;   // 57344
    const long bsG = (long)HEXP * WW;    // 32768

    // ---- stem: fc + 2x (conv3 -> LN -> ReLU) ----
    gemm_wmma_kernel<float, 1><<<dim3(CCH / 64, TT / 16, NB), 128, 0, stream>>>(
        fcw_h, x_in, P(4), XA, CCH, TT, 2048, TT, (long)2048 * TT, TT, bsX);

    pack_conv_kernel<<<(NB * 1536 * TT) / 256, 256, 0, stream>>>(XA, colbuf);
    gemm_wmma_kernel<_Float16, 0><<<dim3(CCH / 64, TT / 16, NB), 128, 0, stream>>>(
        conv_h0, colbuf, nullptr, XB, CCH, TT, 1536, TT, (long)1536 * TT, TT, bsX);
    ln_kernel<<<dim3(TT, NB), 128, 0, stream>>>(XB, XA, P(151), P(150), TT, bsX, TT, bsX, 1);

    pack_conv_kernel<<<(NB * 1536 * TT) / 256, 256, 0, stream>>>(XA, colbuf);
    gemm_wmma_kernel<_Float16, 0><<<dim3(CCH / 64, TT / 16, NB), 128, 0, stream>>>(
        conv_h1, colbuf, nullptr, XB, CCH, TT, 1536, TT, (long)1536 * TT, TT, bsX);
    float* X = (float*)d_out;  // main activation buffer == final concatenated output
    ln_kernel<<<dim3(TT, NB), 128, 0, stream>>>(XB, X, P(153), P(152), TT, bsX, TT, bsX, 1);

    // ---- windowed encoder stack ----
    const dim3 gProj(CCH / 64, WW / 16, NB);
    for (int wi = 0; wi < 8; wi++) {
        float* cur = X + wi * WW;     // column slice, ld = TT
        const int Lh = WW * wi;
        for (int li = 0; li < 8; li++) {
            // --- attention branch ---
            ln_kernel<<<dim3(WW, NB), 128, 0, stream>>>(
                cur, bl, L(li, L_LN1G), L(li, L_LN1B), TT, bsX, WW, bsW, 0);
            if (Lh > 0) {
                ln_kernel<<<dim3(Lh, NB), 128, 0, stream>>>(
                    hist[li], hln, L(li, L_LN1G), L(li, L_LN1B), MEMC, bsH, MEMC, bsH, 0);
                gemm_wmma_kernel<float, 1><<<dim3(CCH / 64, Lh / 16, NB), 128, 0, stream>>>(
                    wk_h[li], hln, L(li, L_BK), khb, CCH, Lh, CCH, MEMC, bsH, MEMC, bsH);
                gemm_wmma_kernel<float, 1><<<dim3(CCH / 64, Lh / 16, NB), 128, 0, stream>>>(
                    wv_h[li], hln, L(li, L_BV), vhb, CCH, Lh, CCH, MEMC, bsH, MEMC, bsH);
            }
            gemm_wmma_kernel<float, 1><<<gProj, 128, 0, stream>>>(
                wq_h[li], bl, L(li, L_BQ), bq_, CCH, WW, CCH, WW, bsW, WW, bsW);
            gemm_wmma_kernel<float, 1><<<gProj, 128, 0, stream>>>(
                wk_h[li], bl, L(li, L_BK), bk_, CCH, WW, CCH, WW, bsW, WW, bsW);
            gemm_wmma_kernel<float, 1><<<gProj, 128, 0, stream>>>(
                wv_h[li], bl, L(li, L_BV), bv_, CCH, WW, CCH, WW, bsW, WW, bsW);

            attn_kernel<<<dim3(NHD, NB), 256, 0, stream>>>(bq_, bk_, bv_, khb, vhb, bo_, Lh);

            gemm_wmma_kernel<float, 1><<<gProj, 128, 0, stream>>>(
                wo_h[li], bo_, L(li, L_BO), bh_, CCH, WW, CCH, WW, bsW, WW, bsW);
            resid_kernel<<<(NB * CCH * WW) / 256, 256, 0, stream>>>(cur, bh_, L(li, L_LS1));

            // --- MLP branch ---
            ln_kernel<<<dim3(WW, NB), 128, 0, stream>>>(
                cur, bl, L(li, L_LN2G), L(li, L_LN2B), TT, bsX, WW, bsW, 0);
            gemm_wmma_kernel<float, 2><<<dim3(HEXP / 64, WW / 16, NB), 128, 0, stream>>>(
                w1_h[li], bl, L(li, L_B1), bg, HEXP, WW, CCH, WW, bsW, WW, bsG);
            gemm_wmma_kernel<float, 1><<<gProj, 128, 0, stream>>>(
                w2_h[li], bg, L(li, L_B2), bh_, CCH, WW, HEXP, WW, bsG, WW, bsW);
            resid_kernel<<<(NB * CCH * WW) / 256, 256, 0, stream>>>(cur, bh_, L(li, L_LS2));

            // --- history append (post-final-window append is never read; merges are dead) ---
            if (wi < 7) {
                append_hist_kernel<<<(NB * CCH * WW) / 256, 256, 0, stream>>>(
                    cur, hist[li], WW * wi);
            }
        }
    }
}
